// MambaLayer_53824530154109
// MI455X (gfx1250) — matrix operand order repaired and checked by reading, not compiled, not run on hardware
//
#include <hip/hip_runtime.h>
#include <hip/hip_bf16.h>

// ---------------------------------------------------------------------------
// Mamba layer for MI455X (gfx1250), wave32.
//  - bf16 WMMA 16x16x32 GEMMs with 32x32 register blocking (4 WMMA / k-step)
//  - chunked parallel selective scan (16 chunks of 128, 3-phase)
// ---------------------------------------------------------------------------

#define DM      512
#define DS      16
#define KCONV   4
#define DI      512
#define DTRANK  32
#define NB      4
#define LSEQ    2048
#define MROWS   (NB * LSEQ)      // 8192 token rows
#define CL      128              // scan chunk length
#define NCH     (LSEQ / CL)      // 16 chunks

typedef __attribute__((ext_vector_type(16))) __bf16 bf16x16;
typedef __attribute__((ext_vector_type(8)))  __bf16 bf16x8;
typedef __attribute__((ext_vector_type(8)))  float  f32x8;

union BF16Frag { bf16x16 v; bf16x8 h[2]; };

// ---------------------------------------------------------------------------
// f32 -> bf16 elementwise conversion
// ---------------------------------------------------------------------------
__global__ void cvt_f32_to_bf16(const float* __restrict__ in,
                                __bf16* __restrict__ out, int n) {
    int i = blockIdx.x * blockDim.x + threadIdx.x;
    if (i < n) out[i] = (__bf16)in[i];
}

// ---------------------------------------------------------------------------
// WMMA GEMM: C[M,N] = A[M,K] (row-major, lda) x W[N,K]^T (row-major, ldw)
// One wave32 computes a 32x32 output block = 2x2 grid of 16x16 WMMA tiles.
//
// Fragment layouts per CDNA5 ISA 7.12.2 (wave32):
//   A (16xK, bf16): lane m(0..15)=row; elems 0..7 -> K=k0+8*half+i,
//                   elems 8..15 -> K=k0+16+8*half+i      (half = lane>=16)
//   B (Kx16, bf16): lane = column N; same K mapping as A
//   C/D (16x16 f32): acc[r] -> row base+r+8*half, col tN+(lane&15)
// ---------------------------------------------------------------------------
enum { EPI_F32 = 0, EPI_F32_BF16 = 1, EPI_SOFTPLUS = 2, EPI_CLIP = 3 };

template <int EPI>
__device__ inline void store_tile(const f32x8& acc, int row0, int col, int half,
                                  float* __restrict__ C, int ldc,
                                  __bf16* __restrict__ Cb,
                                  const float* __restrict__ bias) {
#pragma unroll
    for (int r = 0; r < 8; ++r) {
        const int row = row0 + r + 8 * half;
        const size_t off = (size_t)row * ldc + col;
        float v = acc[r];
        if (EPI == EPI_SOFTPLUS) {
            v += bias[col];
            v = (v > 20.0f) ? v : log1pf(__expf(v));
            C[off] = v;
        } else if (EPI == EPI_CLIP) {
            v = (v != v) ? 0.0f : v;                 // NaN -> 0
            v = fminf(fmaxf(v, -1000.0f), 1000.0f);  // clip
            C[off] = v;
        } else {
            C[off] = v;
            if (EPI == EPI_F32_BF16) Cb[off] = (__bf16)v;
        }
    }
}

template <int EPI>
__global__ __launch_bounds__(256) void gemm_wmma_bf16(
    const __bf16* __restrict__ A, int lda,
    const __bf16* __restrict__ W, int ldw,
    float* __restrict__ C, int ldc,
    __bf16* __restrict__ Cb,              // secondary bf16 output (EPI_F32_BF16)
    const float* __restrict__ bias,       // softplus bias (EPI_SOFTPLUS)
    int N, int Kdim)
{
    const int lane  = threadIdx.x & 31;
    const int wave  = threadIdx.x >> 5;
    const int tile  = blockIdx.x * 8 + wave;
    const int tilesN = N >> 5;                  // 32-wide output blocks
    const int tM = (tile / tilesN) << 5;
    const int tN = (tile % tilesN) << 5;
    const int half = lane >> 4;
    const int lid  = lane & 15;

    const __bf16* arow0 = A + (size_t)(tM + lid) * lda;
    const __bf16* arow1 = arow0 + (size_t)16 * lda;
    const __bf16* wrow0 = W + (size_t)(tN + lid) * ldw;
    const __bf16* wrow1 = wrow0 + (size_t)16 * ldw;

    f32x8 acc00 = {}, acc01 = {}, acc10 = {}, acc11 = {};
    for (int k0 = 0; k0 < Kdim; k0 += 32) {
        BF16Frag a0, a1, b0, b1;
        a0.h[0] = *(const bf16x8*)(arow0 + k0 + 8 * half);
        a0.h[1] = *(const bf16x8*)(arow0 + k0 + 16 + 8 * half);
        a1.h[0] = *(const bf16x8*)(arow1 + k0 + 8 * half);
        a1.h[1] = *(const bf16x8*)(arow1 + k0 + 16 + 8 * half);
        b0.h[0] = *(const bf16x8*)(wrow0 + k0 + 8 * half);
        b0.h[1] = *(const bf16x8*)(wrow0 + k0 + 16 + 8 * half);
        b1.h[0] = *(const bf16x8*)(wrow1 + k0 + 8 * half);
        b1.h[1] = *(const bf16x8*)(wrow1 + k0 + 16 + 8 * half);
        acc00 = __builtin_amdgcn_wmma_f32_16x16x32_bf16(false, a0.v, false, b0.v,
                                                        (short)0, acc00, false, false);
        acc01 = __builtin_amdgcn_wmma_f32_16x16x32_bf16(false, a0.v, false, b1.v,
                                                        (short)0, acc01, false, false);
        acc10 = __builtin_amdgcn_wmma_f32_16x16x32_bf16(false, a1.v, false, b0.v,
                                                        (short)0, acc10, false, false);
        acc11 = __builtin_amdgcn_wmma_f32_16x16x32_bf16(false, a1.v, false, b1.v,
                                                        (short)0, acc11, false, false);
    }

    store_tile<EPI>(acc00, tM,      tN + lid,      half, C, ldc, Cb, bias);
    store_tile<EPI>(acc01, tM,      tN + 16 + lid, half, C, ldc, Cb, bias);
    store_tile<EPI>(acc10, tM + 16, tN + lid,      half, C, ldc, Cb, bias);
    store_tile<EPI>(acc11, tM + 16, tN + 16 + lid, half, C, ldc, Cb, bias);
}

// ---------------------------------------------------------------------------
// Depthwise causal conv (K=4) + bias + SiLU.
// ---------------------------------------------------------------------------
__global__ void conv_silu(const float* __restrict__ xz,
                          const float* __restrict__ w,   // (DI, 1, KCONV)
                          const float* __restrict__ bias,
                          float* __restrict__ xc, __bf16* __restrict__ xcb) {
    const int idx = blockIdx.x * blockDim.x + threadIdx.x; // over B*L*DI
    const int d = idx & (DI - 1);
    const int l = (idx >> 9) & (LSEQ - 1);
    const int b = idx >> 20;
    float v = bias[d];
#pragma unroll
    for (int k = 0; k < KCONV; ++k) {
        const int ll = l + k - (KCONV - 1);
        if (ll >= 0)
            v += w[d * KCONV + k] * xz[(size_t)(b * LSEQ + ll) * (2 * DI) + d];
    }
    const float s = v / (1.0f + __expf(-v));   // silu
    xc[idx]  = s;
    xcb[idx] = (__bf16)s;
}

// ---------------------------------------------------------------------------
// Chunked selective scan.  h[l] = exp(dt*a)*h[l-1] + dt*B*xc  is associative,
// so: phase A computes each chunk's end-state from h_in=0 plus sum(dt);
// combine propagates true chunk-incoming states (chunk decay = exp(a*sum dt));
// phase B replays each chunk from the true h_in and emits gated y (bf16).
// One lane per (b,d[,chunk]) channel; DS=16 states in registers.
// ---------------------------------------------------------------------------
__global__ __launch_bounds__(256) void scan_partial(
    const float* __restrict__ dtf,    // (B,L,DI)
    const float* __restrict__ xc,     // (B,L,DI)
    const float* __restrict__ xdbl,   // (B,L,64): B at [32,48)
    const float* __restrict__ A_log,  // (DI,DS)
    float* __restrict__ hend,         // (NCH,B,DS,DI)
    float* __restrict__ dts)          // (NCH,B,DI)
{
    const int c = blockIdx.x / (NB * 2);
    const int rem = blockIdx.x % (NB * 2);
    const int b = rem >> 1;
    const int d = (rem & 1) * 256 + threadIdx.x;

    float a[DS], h[DS];
#pragma unroll
    for (int s = 0; s < DS; ++s) {
        a[s] = -__expf(A_log[d * DS + s]);
        h[s] = 0.0f;
    }
    float dtsum = 0.0f;
    for (int l = c * CL; l < (c + 1) * CL; ++l) {
        const size_t row = (size_t)(b * LSEQ + l);
        const float dt = dtf[row * DI + d];
        const float xv = xc[row * DI + d];
        const float* bv = xdbl + row * 64 + DTRANK;
        const float dtx = dt * xv;
        dtsum += dt;
#pragma unroll
        for (int s = 0; s < DS; ++s)
            h[s] = __expf(dt * a[s]) * h[s] + dtx * bv[s];
    }
    const size_t cb = (size_t)(c * NB + b);
#pragma unroll
    for (int s = 0; s < DS; ++s)
        hend[(cb * DS + s) * DI + d] = h[s];
    dts[cb * DI + d] = dtsum;
}

__global__ __launch_bounds__(256) void scan_combine(
    const float* __restrict__ hend,   // (NCH,B,DS,DI)
    const float* __restrict__ dts,    // (NCH,B,DI)
    const float* __restrict__ A_log,  // (DI,DS)
    float* __restrict__ hin)          // (NCH,B,DS,DI)
{
    const int b = blockIdx.x >> 1;
    const int d = (blockIdx.x & 1) * 256 + threadIdx.x;

    float a[DS], h[DS];
#pragma unroll
    for (int s = 0; s < DS; ++s) {
        a[s] = -__expf(A_log[d * DS + s]);
        h[s] = 0.0f;
    }
    for (int c = 0; c < NCH; ++c) {
        const size_t cb = (size_t)(c * NB + b);
        const float dtsum = dts[cb * DI + d];
#pragma unroll
        for (int s = 0; s < DS; ++s) {
            hin[(cb * DS + s) * DI + d] = h[s];              // incoming state
            h[s] = __expf(a[s] * dtsum) * h[s]               // chunk decay
                   + hend[(cb * DS + s) * DI + d];
        }
    }
}

__global__ __launch_bounds__(256) void scan_final(
    const float* __restrict__ dtf,    // (B,L,DI)
    const float* __restrict__ xc,     // (B,L,DI)
    const float* __restrict__ xz,     // (B,L,2*DI): z at cols [DI, 2*DI)
    const float* __restrict__ xdbl,   // (B,L,64): B at [32,48), C at [48,64)
    const float* __restrict__ A_log,  // (DI,DS)
    const float* __restrict__ Dp,     // (DI)
    const float* __restrict__ hin,    // (NCH,B,DS,DI)
    __bf16* __restrict__ yb)          // (B,L,DI)
{
    const int c = blockIdx.x / (NB * 2);
    const int rem = blockIdx.x % (NB * 2);
    const int b = rem >> 1;
    const int d = (rem & 1) * 256 + threadIdx.x;

    float a[DS], h[DS];
    const size_t cb = (size_t)(c * NB + b);
#pragma unroll
    for (int s = 0; s < DS; ++s) {
        a[s] = -__expf(A_log[d * DS + s]);
        h[s] = hin[(cb * DS + s) * DI + d];
    }
    const float Dv = Dp[d];

    for (int l = c * CL; l < (c + 1) * CL; ++l) {
        const size_t row = (size_t)(b * LSEQ + l);
        const float dt = dtf[row * DI + d];
        const float xv = xc[row * DI + d];
        const float zv = xz[row * (2 * DI) + DI + d];
        const float* bc = xdbl + row * 64 + DTRANK;  // B then C, 16 each
        const float dtx = dt * xv;
        float y = 0.0f;
#pragma unroll
        for (int s = 0; s < DS; ++s) {
            h[s] = __expf(dt * a[s]) * h[s] + dtx * bc[s];
            y += h[s] * bc[DS + s];
        }
        y += xv * Dv;
        y *= zv / (1.0f + __expf(-zv));   // * silu(z)
        yb[row * DI + d] = (__bf16)y;
    }
}

// ---------------------------------------------------------------------------
extern "C" void kernel_launch(void* const* d_in, const int* in_sizes, int n_in,
                              void* d_out, int out_size, void* d_ws, size_t ws_size,
                              hipStream_t stream) {
    const float* x       = (const float*)d_in[0];
    const float* W_in    = (const float*)d_in[1];
    const float* conv_w  = (const float*)d_in[2];
    const float* conv_b  = (const float*)d_in[3];
    const float* W_xproj = (const float*)d_in[4];
    const float* W_dt    = (const float*)d_in[5];
    const float* b_dt    = (const float*)d_in[6];
    const float* A_log   = (const float*)d_in[7];
    const float* D_par   = (const float*)d_in[8];
    const float* W_out   = (const float*)d_in[9];
    float* out = (float*)d_out;

    char* ws = (char*)d_ws;
    size_t off = 0;
    auto alloc = [&](size_t bytes) -> void* {
        void* p = ws + off;
        off = (off + bytes + 255) & ~(size_t)255;
        return p;
    };
    __bf16* x_b    = (__bf16*)alloc((size_t)MROWS * DM * 2);
    __bf16* Win_b  = (__bf16*)alloc((size_t)(2 * DI) * DM * 2);
    __bf16* Wxp_b  = (__bf16*)alloc((size_t)(DTRANK + 2 * DS) * DI * 2);
    __bf16* Wdt_b  = (__bf16*)alloc((size_t)DI * DTRANK * 2);
    __bf16* Wout_b = (__bf16*)alloc((size_t)DM * DI * 2);
    float*  xz     = (float*) alloc((size_t)MROWS * (2 * DI) * 4);
    float*  xc     = (float*) alloc((size_t)MROWS * DI * 4);
    __bf16* xc_b   = (__bf16*)alloc((size_t)MROWS * DI * 2);
    float*  xdbl   = (float*) alloc((size_t)MROWS * 64 * 4);
    __bf16* xdbl_b = (__bf16*)alloc((size_t)MROWS * 64 * 2);
    float*  dtf    = (float*) alloc((size_t)MROWS * DI * 4);
    __bf16* y_b    = (__bf16*)alloc((size_t)MROWS * DI * 2);
    float*  hend   = (float*) alloc((size_t)NCH * NB * DS * DI * 4);
    float*  hin    = (float*) alloc((size_t)NCH * NB * DS * DI * 4);
    float*  dts    = (float*) alloc((size_t)NCH * NB * DI * 4);

    // --- bf16 conversions (x + weights) ---
    auto cvt = [&](const float* src, __bf16* dst, int n) {
        cvt_f32_to_bf16<<<(n + 255) / 256, 256, 0, stream>>>(src, dst, n);
    };
    cvt(x,       x_b,    MROWS * DM);
    cvt(W_in,    Win_b,  2 * DI * DM);
    cvt(W_xproj, Wxp_b,  (DTRANK + 2 * DS) * DI);
    cvt(W_dt,    Wdt_b,  DI * DTRANK);
    cvt(W_out,   Wout_b, DM * DI);

    // --- GEMM1: xz = x @ W_in^T  (8192 x 1024, K=512) ---
    gemm_wmma_bf16<EPI_F32><<<(MROWS / 32) * (2 * DI / 32) / 8, 256, 0, stream>>>(
        x_b, DM, Win_b, DM, xz, 2 * DI, nullptr, nullptr, 2 * DI, DM);

    // --- conv + SiLU ---
    conv_silu<<<(MROWS * DI) / 256, 256, 0, stream>>>(xz, conv_w, conv_b, xc, xc_b);

    // --- GEMM2: x_dbl = xc @ W_xproj^T  (8192 x 64, K=512) ---
    gemm_wmma_bf16<EPI_F32_BF16><<<(MROWS / 32) * (64 / 32) / 8, 256, 0, stream>>>(
        xc_b, DI, Wxp_b, DI, xdbl, 64, xdbl_b, nullptr, 64, DI);

    // --- GEMM3: dt = softplus(x_dbl[:, :32] @ W_dt^T + b_dt)  (8192 x 512, K=32) ---
    gemm_wmma_bf16<EPI_SOFTPLUS><<<(MROWS / 32) * (DI / 32) / 8, 256, 0, stream>>>(
        xdbl_b, 64, Wdt_b, DTRANK, dtf, DI, nullptr, b_dt, DI, DTRANK);

    // --- chunked selective scan ---
    scan_partial<<<NCH * NB * (DI / 256), 256, 0, stream>>>(dtf, xc, xdbl, A_log, hend, dts);
    scan_combine<<<NB * (DI / 256), 256, 0, stream>>>(hend, dts, A_log, hin);
    scan_final<<<NCH * NB * (DI / 256), 256, 0, stream>>>(dtf, xc, xz, xdbl, A_log, D_par,
                                                          hin, y_b);

    // --- GEMM4: out = clip(y @ W_out^T)  (8192 x 512, K=512) ---
    gemm_wmma_bf16<EPI_CLIP><<<(MROWS / 32) * (DI / 32) / 8, 256, 0, stream>>>(
        y_b, DI, Wout_b, DI, out, DI, nullptr, nullptr, DI, DI);

    (void)in_sizes; (void)n_in; (void)out_size; (void)ws_size;
}
